// CouplingLayer_46712064312170
// MI455X (gfx1250) — compile-verified, matrix operand order checked by reading, not run on hardware
//
#include <hip/hip_runtime.h>
#include <hip/hip_fp16.h>
#include <cstddef>

// ---------------------------------------------------------------------------
// Types for WMMA fragments (gfx1250, wave32)
// ---------------------------------------------------------------------------
typedef __attribute__((ext_vector_type(16))) _Float16 v16h;
typedef __attribute__((ext_vector_type(8)))  _Float16 v8h;
typedef __attribute__((ext_vector_type(8)))  float    v8f;

union Frag16 { v16h v; v8h h[2]; };

__device__ __forceinline__ v8h zero_v8h() {
  v8h z;
#pragma unroll
  for (int i = 0; i < 8; ++i) z[i] = (_Float16)0.0f;
  return z;
}
__device__ __forceinline__ v8f zero_v8f() {
  v8f z;
#pragma unroll
  for (int i = 0; i < 8; ++i) z[i] = 0.0f;
  return z;
}
__device__ __forceinline__ float elu_f(float v) {
  return v > 0.0f ? v : expm1f(v);
}

// ---------------------------------------------------------------------------
// Problem constants
// ---------------------------------------------------------------------------
#define BB   16
#define HH   64
#define WW   64
#define C1C  8
#define C2C  8
#define HID  128
#define OUTC 72      // C2*(C2+1)
#define NPIX (BB*HH*WW)          // 65536
#define OUT_IMG (BB*16*HH*WW)    // 1048576 floats in d_out before log_det

// ---------------------------------------------------------------------------
// Prep: x (f32 NCHW, 16ch) -> x1 (f16 NHWC, 8ch)
// ---------------------------------------------------------------------------
__global__ void k_prep_x(const float* __restrict__ x, _Float16* __restrict__ dst) {
  int idx = blockIdx.x * 256 + threadIdx.x;          // over NPIX*8
  if (idx >= NPIX * C1C) return;
  int c  = idx & 7;
  int p  = idx >> 3;            // b*4096 + h*64 + w
  int b  = p >> 12;
  int hw = p & 4095;
  dst[idx] = (_Float16)x[((size_t)b * 16 + c) * 4096 + hw];
}

// ---------------------------------------------------------------------------
// Prep: weights OIHW f32 -> f16 [COP][TAPP*CIN], scale folded, zero padded
// ---------------------------------------------------------------------------
__global__ void k_prep_w(const float* __restrict__ w, const float* __restrict__ s,
                         _Float16* __restrict__ dst,
                         int CO, int COP, int CIN, int KS, int TAPP) {
  int ktot = TAPP * CIN;
  int idx = blockIdx.x * 256 + threadIdx.x;
  if (idx >= COP * ktot) return;
  int co  = idx / ktot;
  int k   = idx - co * ktot;
  int tap = k / CIN;
  int ci  = k - tap * CIN;
  float v = 0.0f;
  if (co < CO && tap < KS * KS) {
    int kh = tap / KS, kw = tap - kh * KS;
    v = w[(((size_t)co * CIN + ci) * KS + kh) * KS + kw];
    if (s) v *= s[co];
  }
  dst[idx] = (_Float16)v;
}

// ---------------------------------------------------------------------------
// Implicit-GEMM conv via v_wmma_f32_16x16x32_f16.
//   in    : f16 NHWC [B][H][W][CINP]
//   wgt   : f16 [COP][TAPP*CINP]   (scale folded in)
//   out_h : f16 NHWC [B][H][W][128]          (when !OUTF32)
//   out_f : f32 NCHW [B][CO][H][W]           (when OUTF32)
// One workgroup = one (b,h) row; 8 waves; wave wv -> channels [16*wv,16*wv+16),
// 4 accumulators covering the 64 pixels of the row.
// ---------------------------------------------------------------------------
template<int CINP, int NWC, int KS, bool ACT, bool RES, bool OUTF32>
__global__ __launch_bounds__(256) void k_conv_wmma(
    const _Float16* __restrict__ in,
    const _Float16* __restrict__ wgt,
    const float*    __restrict__ bias,
    const _Float16* __restrict__ res,
    _Float16*       __restrict__ out_h,
    float*          __restrict__ out_f,
    const int CO)
{
  constexpr int PAD   = (KS == 3) ? 1 : 0;
  constexpr int ROWS  = (KS == 3) ? 3 : 1;
  constexpr int TAPP  = (KS == 3) ? ((CINP == 8) ? 12 : 9) : 1;  // pad K to /32
  constexpr int KTOTP = TAPP * CINP;
  constexpr int CPP   = CINP / 8;        // 16B chunks per pixel

  __shared__ alignas(16) _Float16 smem[ROWS * WW * CINP];

  const int bh  = blockIdx.x;            // b*64 + h
  const int b   = bh >> 6;
  const int h   = bh & 63;
  const int tid = threadIdx.x;

  // ---- stage input rows (h-PAD .. h+PAD) into LDS, zero-fill OOB rows ----
  const int nchunk = ROWS * WW * CPP;
  for (int q = tid; q < nchunk; q += 256) {
    const int r   = q / (WW * CPP);
    const int rem = q - r * (WW * CPP);
    const int wpx = rem / CPP;
    const int c0  = (rem - wpx * CPP) * 8;
    const int hr  = h + r - PAD;
    v8h val = zero_v8h();
    if (hr >= 0 && hr < HH)
      val = *(const v8h*)(in + (((size_t)(b * HH + hr) * WW + wpx) * CINP + c0));
    *(v8h*)(smem + ((size_t)(r * WW + wpx) * CINP + c0)) = val;
  }
  __syncthreads();

  const int wv   = tid >> 5;
  const int lane = tid & 31;
  if (wv >= NWC) return;
  const int co0 = wv * 16;
  const int lg  = lane >> 4;   // lane group (0: lanes 0-15, 1: lanes 16-31)
  const int lm  = lane & 15;

  v8f acc[4];
#pragma unroll
  for (int s = 0; s < 4; ++s) acc[s] = zero_v8f();

  const _Float16* wrow = wgt + (size_t)(co0 + lm) * KTOTP;

  for (int ko = 0; ko < KTOTP; ko += 32) {
    // A fragment: row co0+lm, contiguous-K chunks per documented 16-bit layout
    Frag16 afrag;
    afrag.h[0] = *(const v8h*)(wrow + ko + 8 * lg);
    afrag.h[1] = *(const v8h*)(wrow + ko + 16 + 8 * lg);
#pragma unroll
    for (int s = 0; s < 4; ++s) {
      Frag16 bfrag;
#pragma unroll
      for (int cc = 0; cc < 2; ++cc) {
        const int k   = ko + 16 * cc + 8 * lg;   // 8-wide K chunk start
        const int tap = k / CINP;                // never splits a chunk
        const int ci  = k - tap * CINP;
        v8h val = zero_v8h();
        if (tap < KS * KS) {
          const int kh = tap / KS;
          const int kw = tap - kh * KS;
          const int ws = 16 * s + lm + kw - PAD;
          if (ws >= 0 && ws < WW)
            val = *(const v8h*)(smem + ((size_t)(kh * WW + ws) * CINP + ci));
        }
        bfrag.h[cc] = val;
      }
      acc[s] = __builtin_amdgcn_wmma_f32_16x16x32_f16(
          false, afrag.v, false, bfrag.v, (short)0, acc[s], false, false);
    }
  }

  // ---- epilogue: bias (+residual) (+ELU), store ----
  const int cobase = co0 + lg * 8;
#pragma unroll
  for (int s = 0; s < 4; ++s) {
    const int wp = 16 * s + lm;
#pragma unroll
    for (int r = 0; r < 8; ++r) {
      const int co = cobase + r;
      if (co < CO) {
        float v = acc[s][r] + bias[co];
        if (RES) {
          v += (float)res[(((size_t)(b * HH + h) * WW + wp) * HID + co)];
          v = elu_f(v);
        } else if (ACT) {
          v = elu_f(v);
        }
        if (OUTF32) {
          out_f[(((size_t)b * CO + co) * HH + h) * WW + wp] = v;
        } else {
          out_h[(((size_t)(b * HH + h) * WW + wp) * HID + co)] = (_Float16)v;
        }
      }
    }
  }
}

// ---------------------------------------------------------------------------
// Per-pixel: build 8x8 M = rescale*tanh(scale*net + shift) + reshift,
// E = expm(M) via column-wise Taylor-14 (||M||inf <= 1), y2 = E x2 + shift,
// write concat([x1, y2]); block-partial trace sums for log_det.
// ---------------------------------------------------------------------------
__global__ __launch_bounds__(256) void k_expm_apply(
    const float* __restrict__ net,     // [16][72][64][64] NCHW
    const float* __restrict__ x,       // [16][16][64][64] NCHW
    const float* __restrict__ s_scale, const float* __restrict__ s_shift,
    const float* __restrict__ s_rescale, const float* __restrict__ s_reshift,
    float* __restrict__ out,           // d_out (image part)
    float* __restrict__ partials)      // [256] block partial traces
{
  const int p  = blockIdx.x * 256 + threadIdx.x;   // 0..65535
  const int b  = p >> 12;
  const int hw = p & 4095;
  const size_t nb = (size_t)b * OUTC * 4096 + hw;  // net base (co stride 4096)
  const float ss = s_scale[0], sh = s_shift[0], rs = s_rescale[0], rh = s_reshift[0];

  float M[8][8];
  float tr = 0.0f;
#pragma unroll
  for (int i = 0; i < 8; ++i) {
#pragma unroll
    for (int c = 0; c < 8; ++c) {
      float raw = net[nb + (size_t)((i + 1) * 8 + c) * 4096];
      float m = rs * tanhf(ss * raw + sh) + rh;
      M[c][i] = m;                                  // wt[b,h,w,c,i]
      if (i == c) tr += m;
    }
  }

  // deterministic block reduction of trace
  __shared__ float red[256];
  red[threadIdx.x] = tr;
  __syncthreads();
  for (int st = 128; st > 0; st >>= 1) {
    if (threadIdx.x < st) red[threadIdx.x] += red[threadIdx.x + st];
    __syncthreads();
  }
  if (threadIdx.x == 0) partials[blockIdx.x] = red[0];

  // x2, shift
  const size_t xb = (size_t)b * 16 * 4096 + hw;
  float x2[8], y2[8];
#pragma unroll
  for (int j = 0; j < 8; ++j) x2[j] = x[xb + (size_t)(8 + j) * 4096];
#pragma unroll
  for (int c = 0; c < 8; ++c) y2[c] = net[nb + (size_t)c * 4096];  // shift

  // E columns by Taylor series; accumulate y2 += E[:,j]*x2[j]
#pragma unroll
  for (int j = 0; j < 8; ++j) {
    float v[8], sc[8];
#pragma unroll
    for (int c = 0; c < 8; ++c) { v[c] = (c == j) ? 1.0f : 0.0f; sc[c] = v[c]; }
    for (int k = 1; k <= 14; ++k) {
      const float inv = 1.0f / (float)k;
      float nv[8];
#pragma unroll
      for (int c = 0; c < 8; ++c) {
        float a = 0.0f;
#pragma unroll
        for (int t = 0; t < 8; ++t) a = fmaf(M[c][t], v[t], a);
        nv[c] = a * inv;
      }
#pragma unroll
      for (int c = 0; c < 8; ++c) { v[c] = nv[c]; sc[c] += nv[c]; }
    }
#pragma unroll
    for (int c = 0; c < 8; ++c) y2[c] = fmaf(sc[c], x2[j], y2[c]);
  }

  // store concat([x1, y2])
#pragma unroll
  for (int c = 0; c < 8; ++c) {
    out[xb + (size_t)c * 4096]       = x[xb + (size_t)c * 4096];
    out[xb + (size_t)(8 + c) * 4096] = y2[c];
  }
}

__global__ void k_logdet_final(const float* __restrict__ partials,
                               float* __restrict__ logdet) {
  int b = threadIdx.x;
  if (b < BB) {
    float s = 0.0f;
    for (int i = 0; i < 16; ++i) s += partials[b * 16 + i];
    logdet[b] = s;
  }
}

// ---------------------------------------------------------------------------
// Host launcher
// ---------------------------------------------------------------------------
extern "C" void kernel_launch(void* const* d_in, const int* in_sizes, int n_in,
                              void* d_out, int out_size, void* d_ws, size_t ws_size,
                              hipStream_t stream) {
  (void)out_size; (void)ws_size; (void)n_in;

  const float *x, *in_w, *in_s, *in_b;
  const float *cw[2][3], *cs[2][3], *cb[2][3];
  const float *out_w, *out_b, *p_scale, *p_shift, *p_rescale, *p_reshift;

  if (in_sizes[0] == OUT_IMG) {
    // insertion order: x, y, in.{w,s,b}, blocks[2]{c1,c2,c3}.{w,s,b},
    //                  out_w, out_b, s_scale, s_shift, s_rescale, s_reshift
    x = (const float*)d_in[0];
    in_w = (const float*)d_in[2]; in_s = (const float*)d_in[3]; in_b = (const float*)d_in[4];
    int k = 5;
    for (int blk = 0; blk < 2; ++blk)
      for (int c = 0; c < 3; ++c) {
        cw[blk][c] = (const float*)d_in[k++];
        cs[blk][c] = (const float*)d_in[k++];
        cb[blk][c] = (const float*)d_in[k++];
      }
    out_w = (const float*)d_in[23]; out_b = (const float*)d_in[24];
    p_scale = (const float*)d_in[25]; p_shift = (const float*)d_in[26];
    p_rescale = (const float*)d_in[27]; p_reshift = (const float*)d_in[28];
  } else {
    // alphabetical pytree order: blocks[2]{c1,c2,c3}.{b,s,w}, in.{b,s,w},
    //   out_b, out_w, s_rescale, s_reshift, s_scale, s_shift, x, y
    int k = 0;
    for (int blk = 0; blk < 2; ++blk)
      for (int c = 0; c < 3; ++c) {
        cb[blk][c] = (const float*)d_in[k++];
        cs[blk][c] = (const float*)d_in[k++];
        cw[blk][c] = (const float*)d_in[k++];
      }
    in_b = (const float*)d_in[18]; in_s = (const float*)d_in[19]; in_w = (const float*)d_in[20];
    out_b = (const float*)d_in[21]; out_w = (const float*)d_in[22];
    p_rescale = (const float*)d_in[23]; p_reshift = (const float*)d_in[24];
    p_scale = (const float*)d_in[25]; p_shift = (const float*)d_in[26];
    x = (const float*)d_in[27];
  }

  // ---- workspace carve-up ----
  char* ws = (char*)d_ws;
  size_t off = 0;
  auto carve = [&](size_t bytes) -> void* {
    void* ptr = ws + off;
    off = (off + bytes + 255) & ~(size_t)255;
    return ptr;
  };
  _Float16* xin    = (_Float16*)carve((size_t)NPIX * C1C * 2);          // 1 MB
  _Float16* actA   = (_Float16*)carve((size_t)NPIX * HID * 2);          // 16 MB
  _Float16* actB   = (_Float16*)carve((size_t)NPIX * HID * 2);
  _Float16* actC   = (_Float16*)carve((size_t)NPIX * HID * 2);
  float*    netout = (float*)   carve((size_t)NPIX * OUTC * 4);         // 18 MB
  _Float16* w_in   = (_Float16*)carve((size_t)HID * 12 * C1C * 2);      // 128x96
  _Float16* w_c[2][3];
  for (int blk = 0; blk < 2; ++blk) {
    w_c[blk][0] = (_Float16*)carve((size_t)HID * 9 * HID * 2);
    w_c[blk][1] = (_Float16*)carve((size_t)HID * 1 * HID * 2);
    w_c[blk][2] = (_Float16*)carve((size_t)HID * 9 * HID * 2);
  }
  _Float16* w_out  = (_Float16*)carve((size_t)80 * 9 * HID * 2);        // 80x1152
  float*    part   = (float*)   carve(256 * 4);

  // ---- prep ----
  k_prep_x<<<(NPIX * C1C + 255) / 256, 256, 0, stream>>>(x, xin);
  k_prep_w<<<(HID * 12 * C1C + 255) / 256, 256, 0, stream>>>(
      in_w, in_s, w_in, HID, HID, C1C, 3, 12);
  for (int blk = 0; blk < 2; ++blk) {
    k_prep_w<<<(HID * 9 * HID + 255) / 256, 256, 0, stream>>>(
        cw[blk][0], cs[blk][0], w_c[blk][0], HID, HID, HID, 3, 9);
    k_prep_w<<<(HID * 1 * HID + 255) / 256, 256, 0, stream>>>(
        cw[blk][1], cs[blk][1], w_c[blk][1], HID, HID, HID, 1, 1);
    k_prep_w<<<(HID * 9 * HID + 255) / 256, 256, 0, stream>>>(
        cw[blk][2], cs[blk][2], w_c[blk][2], HID, HID, HID, 3, 9);
  }
  k_prep_w<<<(80 * 9 * HID + 255) / 256, 256, 0, stream>>>(
      out_w, nullptr, w_out, OUTC, 80, HID, 3, 9);

  const dim3 cgrid(BB * HH), cblk(256);
  // in conv: 8 -> 128, 3x3, scale/bias + ELU
  k_conv_wmma<8, 8, 3, true, false, false><<<cgrid, cblk, 0, stream>>>(
      xin, w_in, in_b, nullptr, actA, nullptr, HID);
  // block 0: A -> B -> C, c3: C (+A) -> B
  k_conv_wmma<128, 8, 3, true, false, false><<<cgrid, cblk, 0, stream>>>(
      actA, w_c[0][0], cb[0][0], nullptr, actB, nullptr, HID);
  k_conv_wmma<128, 8, 1, true, false, false><<<cgrid, cblk, 0, stream>>>(
      actB, w_c[0][1], cb[0][1], nullptr, actC, nullptr, HID);
  k_conv_wmma<128, 8, 3, false, true, false><<<cgrid, cblk, 0, stream>>>(
      actC, w_c[0][2], cb[0][2], actA, actB, nullptr, HID);
  // block 1: B -> C -> A, c3: A (+B) -> C
  k_conv_wmma<128, 8, 3, true, false, false><<<cgrid, cblk, 0, stream>>>(
      actB, w_c[1][0], cb[1][0], nullptr, actC, nullptr, HID);
  k_conv_wmma<128, 8, 1, true, false, false><<<cgrid, cblk, 0, stream>>>(
      actC, w_c[1][1], cb[1][1], nullptr, actA, nullptr, HID);
  k_conv_wmma<128, 8, 3, false, true, false><<<cgrid, cblk, 0, stream>>>(
      actA, w_c[1][2], cb[1][2], actB, actC, nullptr, HID);
  // out conv: 128 -> 72 (padded to 80 rows, 5 compute waves), f32 NCHW out
  k_conv_wmma<128, 5, 3, false, false, true><<<cgrid, cblk, 0, stream>>>(
      actC, w_out, out_b, nullptr, nullptr, netout, OUTC);

  // expm + einsum + shift + concat + trace partials
  k_expm_apply<<<NPIX / 256, 256, 0, stream>>>(
      netout, x, p_scale, p_shift, p_rescale, p_reshift,
      (float*)d_out, part);
  k_logdet_final<<<1, 16, 0, stream>>>(part, (float*)d_out + OUT_IMG);
}
